// Head_1451698946167
// MI455X (gfx1250) — matrix operand order repaired and checked by reading, not compiled, not run on hardware
//
#include <hip/hip_runtime.h>

typedef __attribute__((ext_vector_type(16))) __bf16 v16bf;
typedef __attribute__((ext_vector_type(8)))  __bf16 v8bf;
typedef __attribute__((ext_vector_type(8)))  float  v8f;

#define B_  4
#define T_  4096
#define C_  512
#define H_  64

static __device__ __forceinline__ v8f wmma_bf16(v16bf a, v16bf b, v8f c) {
  // D = A(16x32 bf16) * B(32x16 bf16) + C(16x16 f32)
  return __builtin_amdgcn_wmma_f32_16x16x32_bf16(false, a, false, b, (short)0, c, false, false);
}

// A-fragment (16x32 bf16) from a row-major bf16 row (lane = row l16, half hf):
// v[0..15] <-> K = base+0..15, base = kchunk*32 + hf*16
static __device__ __forceinline__ v16bf load_afrag(const __bf16* base) {
  v8bf lo = *(const v8bf*)(base);
  v8bf hi = *(const v8bf*)(base + 8);
  v16bf r;
#pragma unroll
  for (int e = 0; e < 8; ++e) { r[e] = lo[e]; r[8 + e] = hi[e]; }
  return r;
}

// ---------------- kernel 1: W -> Wt (bf16, transposed [3][64][512]) ----------------
// Wq gets 1/sqrt(C) folded in so attention scores need no scaling.
__global__ void wt_kernel(const float* __restrict__ Wk, const float* __restrict__ Wq,
                          const float* __restrict__ Wv, __bf16* __restrict__ Wt) {
  int idx = blockIdx.x * 256 + threadIdx.x;      // idx = mat*64*512 + h*512 + c
  if (idx >= 3 * H_ * C_) return;
  int mat = idx / (H_ * C_);
  int rem = idx % (H_ * C_);
  int h = rem / C_;
  int c = rem % C_;
  const float* W = (mat == 0) ? Wk : ((mat == 1) ? Wq : Wv);
  float v = W[c * H_ + h];
  if (mat == 1) v *= 0.04419417382415922f;       // 1/sqrt(512)
  Wt[idx] = (__bf16)v;
}

// ---------------- kernel 2: projections K,Q,V via WMMA ----------------
// Qb/Kb: [B*T, 64] bf16 row-major.  Vt: [B*64, T] bf16 (transposed).
__global__ __launch_bounds__(128) void proj_kernel(const float* __restrict__ x,
                                                   const __bf16* __restrict__ Wt,
                                                   __bf16* __restrict__ Kb,
                                                   __bf16* __restrict__ Qb,
                                                   __bf16* __restrict__ Vt) {
  const int wave = threadIdx.x >> 5;
  const int lane = threadIdx.x & 31;
  const int tile = blockIdx.x * 4 + wave;        // 1024 tiles total
  const int b  = tile >> 8;                      // 256 tiles per batch
  const int t0 = (tile & 255) << 4;
  const int l16 = lane & 15;
  const int hf  = lane >> 4;

  v8f acc[3][4] = {};
  const float* xrow = x + ((size_t)(b * T_ + t0 + l16)) * C_;

  for (int ks = 0; ks < C_ / 32; ++ks) {
    // issue all 12 B-fragment loads first (distinct regs -> overlapped latency)
    v16bf bw[12];
#pragma unroll
    for (int mat = 0; mat < 3; ++mat)
#pragma unroll
      for (int nt = 0; nt < 4; ++nt)
        bw[mat * 4 + nt] = *(const v16bf*)(Wt + ((size_t)(mat * H_ + nt * 16 + l16)) * C_ +
                                           ks * 32 + hf * 16);
    // A-fragment of x (f32 -> bf16 inline)
    const float* p = xrow + ks * 32 + hf * 16;
    v16bf a;
#pragma unroll
    for (int e = 0; e < 8; ++e) { a[e] = (__bf16)p[e]; a[8 + e] = (__bf16)p[8 + e]; }
#pragma unroll
    for (int mat = 0; mat < 3; ++mat)
#pragma unroll
      for (int nt = 0; nt < 4; ++nt)
        acc[mat][nt] = wmma_bf16(a, bw[mat * 4 + nt], acc[mat][nt]);
  }

  // scatter D-layout (lane: rows r+8*hf, col l16) to bf16 outputs
#pragma unroll
  for (int nt = 0; nt < 4; ++nt)
#pragma unroll
    for (int r = 0; r < 8; ++r) {
      int m = r + hf * 8;
      int hcol = nt * 16 + l16;
      size_t row = (size_t)(b * T_ + t0 + m);
      Kb[row * H_ + hcol] = (__bf16)acc[0][nt][r];
      Qb[row * H_ + hcol] = (__bf16)acc[1][nt][r];
      Vt[((size_t)(b * H_ + hcol)) * T_ + (t0 + m)] = (__bf16)acc[2][nt][r];
    }
}

// ---------------- kernel 3: causal flash attention, 64-key blocks ----------------
__global__ __launch_bounds__(128) void attn_kernel(const __bf16* __restrict__ Qb,
                                                   const __bf16* __restrict__ Kb,
                                                   const __bf16* __restrict__ Vt,
                                                   float* __restrict__ out) {
  __shared__ __bf16 Plds[4][16 * 64];            // 2KB per wave
  const int wave = threadIdx.x >> 5;
  const int lane = threadIdx.x & 31;
  const int tile = blockIdx.x * 4 + wave;
  const int b  = tile >> 8;
  const int q0 = (tile & 255) << 4;
  const int l16 = lane & 15;
  const int hf  = lane >> 4;

  // Q A-fragments (scale already folded into Wq)
  const __bf16* qbase = Qb + ((size_t)(b * T_ + q0 + l16)) * H_;
  v16bf aq0 = load_afrag(qbase + 0 * 32 + hf * 16);
  v16bf aq1 = load_afrag(qbase + 1 * 32 + hf * 16);

  // constant all-ones B fragment: P x 1 puts the row-sum in EVERY lane's accumulator
  v16bf ones;
#pragma unroll
  for (int e = 0; e < 16; ++e) ones[e] = (__bf16)1.0f;

  v8f o0 = {}, o1 = {}, o2 = {}, o3 = {}, oL = {};   // oL accumulates softmax denominator
  float m8[8];
#pragma unroll
  for (int r = 0; r < 8; ++r) m8[r] = -__builtin_inff();

  const int kb_end = (q0 + 15) >> 6;             // last 64-key block (inclusive)
  for (int kb = 0; kb <= kb_end; ++kb) {
    const int k0 = kb << 6;

    // ---- issue K fragments (4 subtiles x 2 kchunks) + V chunk0 fragments ----
    v16bf bk[4][2];
#pragma unroll
    for (int sub = 0; sub < 4; ++sub) {
      const __bf16* kr = Kb + ((size_t)(b * T_ + k0 + 16 * sub + l16)) * H_;
      bk[sub][0] = *(const v16bf*)(kr + hf * 16);
      bk[sub][1] = *(const v16bf*)(kr + 32 + hf * 16);
    }
    v16bf bv0[4];
#pragma unroll
    for (int t = 0; t < 4; ++t)
      bv0[t] = *(const v16bf*)(Vt + ((size_t)(b * H_ + t * 16 + l16)) * T_ + k0 + hf * 16);

    // prefetch next block's K/V cachelines (128B rows) -> global_prefetch
    if (kb < kb_end) {
      const int kn = k0 + 64;
#pragma unroll
      for (int sub = 0; sub < 4; ++sub)
        __builtin_prefetch(Kb + ((size_t)(b * T_ + kn + 16 * sub + l16)) * H_, 0, 0);
#pragma unroll
      for (int t = 0; t < 4; ++t)
        __builtin_prefetch(Vt + ((size_t)(b * H_ + t * 16 + l16)) * T_ + kn, 0, 0);
    }

    // ---- S = Q K^T (16 x 64) ----
    v8f s[4];
#pragma unroll
    for (int sub = 0; sub < 4; ++sub) {
      v8f z = {};
      z = wmma_bf16(aq0, bk[sub][0], z);
      s[sub] = wmma_bf16(aq1, bk[sub][1], z);
    }

    // ---- issue V chunk1 fragments; latency hidden behind softmax ----
    v16bf bv1[4];
#pragma unroll
    for (int t = 0; t < 4; ++t)
      bv1[t] = *(const v16bf*)(Vt + ((size_t)(b * H_ + t * 16 + l16)) * T_ + k0 + 32 + hf * 16);

    const bool diag = (k0 + 63 > q0);

    // ---- online softmax: only the row-max needs cross-lane shuffles ----
#pragma unroll
    for (int r = 0; r < 8; ++r) {
      const int m = r + hf * 8;
      float sc0 = s[0][r], sc1 = s[1][r], sc2 = s[2][r], sc3 = s[3][r];
      if (diag) {
        if (k0 +  0 + l16 > q0 + m) sc0 = -__builtin_inff();
        if (k0 + 16 + l16 > q0 + m) sc1 = -__builtin_inff();
        if (k0 + 32 + l16 > q0 + m) sc2 = -__builtin_inff();
        if (k0 + 48 + l16 > q0 + m) sc3 = -__builtin_inff();
      }
      float bm = fmaxf(fmaxf(sc0, sc1), fmaxf(sc2, sc3));
#pragma unroll
      for (int off = 8; off >= 1; off >>= 1) bm = fmaxf(bm, __shfl_xor(bm, off, 32));
      const float mn = fmaxf(m8[r], bm);
      const float al = __expf(m8[r] - mn);
      m8[r] = mn;
      const float e0 = __expf(sc0 - mn);
      const float e1 = __expf(sc1 - mn);
      const float e2 = __expf(sc2 - mn);
      const float e3 = __expf(sc3 - mn);
      o0[r] *= al; o1[r] *= al; o2[r] *= al; o3[r] *= al; oL[r] *= al;
      __bf16* pr = &Plds[wave][m * 64 + l16];
      pr[0]  = (__bf16)e0;
      pr[16] = (__bf16)e1;
      pr[32] = (__bf16)e2;
      pr[48] = (__bf16)e3;
    }

    // ---- P A-fragments (two 16x32 chunks) ----
    v16bf ap0 = load_afrag(&Plds[wave][l16 * 64 +  0 + hf * 16]);
    v16bf ap1 = load_afrag(&Plds[wave][l16 * 64 + 32 + hf * 16]);

    // ---- O += P V ; denominator via ones-column ----
    o0 = wmma_bf16(ap0, bv0[0], o0);
    o1 = wmma_bf16(ap0, bv0[1], o1);
    o2 = wmma_bf16(ap0, bv0[2], o2);
    o3 = wmma_bf16(ap0, bv0[3], o3);
    oL = wmma_bf16(ap0, ones,   oL);
    o0 = wmma_bf16(ap1, bv1[0], o0);
    o1 = wmma_bf16(ap1, bv1[1], o1);
    o2 = wmma_bf16(ap1, bv1[2], o2);
    o3 = wmma_bf16(ap1, bv1[3], o3);
    oL = wmma_bf16(ap1, ones,   oL);
  }

  // epilogue: every lane already holds its rows' denominator in oL
#pragma unroll
  for (int r = 0; r < 8; ++r) {
    const int m = r + hf * 8;
    const float inv = 1.0f / oL[r];
    float* orow = out + ((size_t)(b * T_ + q0 + m)) * H_;
    orow[0  + l16] = o0[r] * inv;
    orow[16 + l16] = o1[r] * inv;
    orow[32 + l16] = o2[r] * inv;
    orow[48 + l16] = o3[r] * inv;
  }
}

extern "C" void kernel_launch(void* const* d_in, const int* in_sizes, int n_in,
                              void* d_out, int out_size, void* d_ws, size_t ws_size,
                              hipStream_t stream) {
  (void)in_sizes; (void)n_in; (void)out_size; (void)ws_size;
  const float* x  = (const float*)d_in[0];
  const float* Wk = (const float*)d_in[1];
  const float* Wq = (const float*)d_in[2];
  const float* Wv = (const float*)d_in[3];
  float* out = (float*)d_out;

  char* ws = (char*)d_ws;
  const size_t WT_BYTES = (size_t)3 * H_ * C_ * 2;          // 196608
  const size_t QK_BYTES = (size_t)B_ * T_ * H_ * 2;         // 2 MB each
  __bf16* Wt = (__bf16*)(ws);
  __bf16* Kb = (__bf16*)(ws + WT_BYTES);
  __bf16* Qb = (__bf16*)(ws + WT_BYTES + QK_BYTES);
  __bf16* Vt = (__bf16*)(ws + WT_BYTES + 2 * QK_BYTES);

  wt_kernel<<<(3 * H_ * C_ + 255) / 256, 256, 0, stream>>>(Wk, Wq, Wv, Wt);
  proj_kernel<<<(B_ * T_ / 16) / 4, 128, 0, stream>>>(x, Wt, Kb, Qb, Vt);
  attn_kernel<<<(B_ * T_ / 16) / 4, 128, 0, stream>>>(Qb, Kb, Vt, out);
}
